// SegLoss_3401614099110
// MI455X (gfx1250) — matrix operand order repaired and checked by reading, not compiled, not run on hardware
//
#include <hip/hip_runtime.h>
#include <hip/hip_bf16.h>

// ---------------- types ----------------
typedef __attribute__((ext_vector_type(16))) _Float16     v16h;
typedef __attribute__((ext_vector_type(8)))  float        v8f;
typedef __attribute__((ext_vector_type(4)))  unsigned int v4u;

union V16H { v16h v; _Float16 h[16]; v4u u[2]; };
union V8F  { v8f  v; float    f[8];  };

// ---------------- problem constants ----------------
#define BB        8
#define KK        64
#define CBASE     4
#define ATTN      14
#define RES       56
#define HF        128
#define WF        128
#define HW        (HF*WF)           // 16384
#define NB        (BB*KK)           // 512  (n)
#define NC        (NB*CBASE)        // 2048 (n*c)
#define NCLASS    80
#define PIX       (RES*RES)         // 3136

// stage-1 GEMM: M1 = NC*ATTN = 28672 rows, K=14 (pad 32), N=56 (pad 64)
#define M1ROWS    (NC*ATTN)         // 28672
#define M1TILES   (M1ROWS/16)       // 1792
// stage-2 GEMM: M2 = NC*RES = 114688 rows, K=14 (pad 32), N=56 (pad 64)
#define M2ROWS    (NC*RES)          // 114688
#define M2TILES   (M2ROWS/16)       // 7168
#define NTILES    4

// ---------------- workspace layout (bytes) ----------------
#define WS_ACC    0                 // 4 floats: [0]=loss acc, [1]=ct sum
#define WS_CTVAL  16                // 512 floats
#define WS_BT     2112              // fp16 B operand image: 4 tiles x 32 lanes x 16 = 4096 B
#define WS_CG     6208              // fp16 gathered coeffs, 28672 x 16 (K-padded, zeros)
#define WS_WT     923712            // fp16 stage-1 out (transposed), 2048 x 56 x 16
#define WS_CUP    4593728           // f32 upsampled coeffs, 2048 x 56(p) x 56(m)

__device__ __forceinline__ float sigmoidf_(float x) { return 1.0f / (1.0f + expf(-x)); }

// ---------------- kernel 0: accumulators + B operand image in WMMA lane layout ----------------
// Bt[t][lane][e]: lane%16 = output column within tile t, e < 8 holds K = 8*(lane/16) + e
// (e >= 8 is K=16..31, all zero-pad). Value = M[N][K], the 56x14
// bilinear interpolation matrix, computed closed-form per entry.
__global__ void k_init(float* acc, _Float16* Bt) {
    if (threadIdx.x == 0) { acc[0] = 0.0f; acc[1] = 0.0f; acc[2] = 0.0f; acc[3] = 0.0f; }
    for (int idx = threadIdx.x; idx < NTILES * 32 * 16; idx += blockDim.x) {
        int t    = idx >> 9;
        int rem  = idx & 511;
        int lane = rem >> 4;
        int e    = rem & 15;
        int N    = t * 16 + (lane & 15);
        float val = 0.0f;
        int k = (lane >> 4) * 8 + e;
        if (e < 8 && N < RES && k < ATTN) {
            float src = ((float)N + 0.5f) * ((float)ATTN / (float)RES) - 0.5f;
            src = fminf(fmaxf(src, 0.0f), (float)(ATTN - 1));
            int i0 = (int)floorf(src);
            int i1 = min(i0 + 1, ATTN - 1);
            float w = src - (float)i0;
            if (k == i0) val += 1.0f - w;
            if (k == i1) val += w;
        }
        Bt[idx] = (_Float16)val;
    }
}

// ---------------- kernel 1: ct_val gather + global sum ----------------
__global__ void k_ctval(const float* __restrict__ center_target,
                        const float* __restrict__ cat_mask,
                        const int*   __restrict__ ind,
                        float* __restrict__ ctval, float* acc) {
    int n = blockIdx.x * blockDim.x + threadIdx.x;
    if (n >= NB) return;
    int b = n >> 6;
    int pos = ind[n];
    const float* ctb = center_target + (size_t)b * NCLASS * HW + pos;
    const float* cm  = cat_mask + (size_t)n * NCLASS;
    float s = 0.0f;
    for (int c = 0; c < NCLASS; ++c) s += ctb[(size_t)c * HW] * cm[c];
    ctval[n] = s;
    atomicAdd(&acc[1], s);
}

// ---------------- kernel 2: gather coefficients -> fp16, 16-half padded rows ----------------
// One thread per A row r = nc*14 + i; writes 14 gathered values + 2 zero pads (32 B).
__global__ void k_gather(const float* __restrict__ shp,
                         const int* __restrict__ ind,
                         _Float16* __restrict__ Cg) {
    int r = blockIdx.x * blockDim.x + threadIdx.x;
    if (r >= M1ROWS) return;
    int nc = r / ATTN;
    int i  = r - nc * ATTN;
    int n  = nc >> 2;
    int c  = nc & 3;
    int b  = n >> 6;
    int pos = ind[n];
    const float* base = shp + (size_t)b * 784 * HW + (size_t)(c * 196 + i * ATTN) * HW + pos;
    _Float16 tmp[16];
#pragma unroll
    for (int j = 0; j < ATTN; ++j) tmp[j] = (_Float16)base[(size_t)j * HW];
    tmp[14] = (_Float16)0.0f;
    tmp[15] = (_Float16)0.0f;
    v4u* dst = (v4u*)(Cg + (size_t)r * 16);
    dst[0] = *(const v4u*)&tmp[0];
    dst[1] = *(const v4u*)&tmp[8];
}

// ---------------- kernel 3: WMMA stage 1  W^T = (C · M^T)^T ----------------
__global__ void k_stage1(const _Float16* __restrict__ Cg,
                         const _Float16* __restrict__ Bt,
                         _Float16* __restrict__ WT) {
    int wave = blockIdx.x * 8 + (threadIdx.x >> 5);   // 7168 waves
    int mt = wave >> 2;
    int nt = wave & 3;
    int lane = threadIdx.x & 31;
    int lrow = lane & 15;
    bool hi = lane >= 16;
    int arow = mt * 16 + lrow;                        // A row < 28672
    int bcol = nt * 16 + lrow;                        // output column p

    const v4u zero4 = {0u, 0u, 0u, 0u};
    V16H A, Bm;
    // A: low half-wave -> K0..7, high half-wave -> K8..15 (one b128); K16..31 = 0
    const v4u* ap = (const v4u*)(Cg + (size_t)arow * 16);
    A.u[0] = ap[hi ? 1 : 0];
    A.u[1] = zero4;
    // B: precomputed lane-layout image, two b128s
    const v4u* bp = (const v4u*)(Bt + ((size_t)nt * 32 + lane) * 16);
    Bm.u[0] = bp[0];
    Bm.u[1] = bp[1];

    V8F D;
#pragma unroll
    for (int e = 0; e < 8; ++e) D.f[e] = 0.0f;
    D.v = __builtin_amdgcn_wmma_f32_16x16x32_f16(false, A.v, false, Bm.v,
                                                 (short)0, D.v, false, false);

    // D[e]: out row = mt*16 + e (+8 hi), out col = bcol; row = nc*14 + i
    if (bcol < RES) {
        int row0 = mt * 16 + (hi ? 8 : 0);
        int nc = row0 / ATTN;
        int i  = row0 - nc * ATTN;
#pragma unroll
        for (int e = 0; e < 8; ++e) {
            WT[(size_t)nc * (RES * 16) + bcol * 16 + i] = (_Float16)D.f[e];
            if (++i == ATTN) { i = 0; ++nc; }
        }
    }
}

// ---------------- kernel 4: WMMA stage 2  U^T = W^T · M^T ----------------
__global__ void k_stage2(const _Float16* __restrict__ WT,
                         const _Float16* __restrict__ Bt,
                         float* __restrict__ CUP) {
    int wave = blockIdx.x * 8 + (threadIdx.x >> 5);   // 28672 waves
    int mt = wave >> 2;
    int nt = wave & 3;
    int lane = threadIdx.x & 31;
    int lrow = lane & 15;
    bool hi = lane >= 16;
    int arow = mt * 16 + lrow;                        // < 114688 (= nc*56 + p)
    int bcol = nt * 16 + lrow;                        // output column m

    const v4u zero4 = {0u, 0u, 0u, 0u};
    V16H A, Bm;
    const v4u* ap = (const v4u*)(WT + (size_t)arow * 16);
    A.u[0] = ap[hi ? 1 : 0];
    A.u[1] = zero4;
    if (hi) {                 // high half-wave elements 6,7 are K=14,15 pads (uninitialized)
        A.h[6] = (_Float16)0.0f;
        A.h[7] = (_Float16)0.0f;
    }
    const v4u* bp = (const v4u*)(Bt + ((size_t)nt * 32 + lane) * 16);
    Bm.u[0] = bp[0];
    Bm.u[1] = bp[1];

    V8F D;
#pragma unroll
    for (int e = 0; e < 8; ++e) D.f[e] = 0.0f;
    D.v = __builtin_amdgcn_wmma_f32_16x16x32_f16(false, A.v, false, Bm.v,
                                                 (short)0, D.v, false, false);

    if (bcol < RES) {
        int row0 = mt * 16 + (hi ? 8 : 0);
        int nc = row0 / RES;
        int p  = row0 - nc * RES;
#pragma unroll
        for (int e = 0; e < 8; ++e) {
            CUP[(size_t)nc * PIX + p * RES + bcol] = D.f[e];   // [nc][p=x][m=y]
            if (++p == RES) { p = 0; ++nc; }
        }
    }
}

// ---------------- kernel 5: fused ROI-align + sigmoid + BCE reduction ----------------
__global__ void k_loss(const float* __restrict__ sal,
                       const float* __restrict__ gtboxes,
                       const float* __restrict__ reg_mask,
                       const float* __restrict__ imask,
                       const float* __restrict__ ctval,
                       const float* __restrict__ CUP,
                       float* acc) {
    int n = blockIdx.x;               // 512 blocks, 256 threads
    int b = n >> 6;
    const float* bx = gtboxes + n * 4;
    float x1 = bx[0] - 0.5f;
    float y1 = bx[1] - 0.5f;
    float bw = (bx[2] - bx[0]) * (1.0f / RES);
    float bh = (bx[3] - bx[1]) * (1.0f / RES);
    const float* salb = sal + (size_t)b * CBASE * HW;
    const float* mk   = imask + (size_t)n * HW;
    const float* cupn = CUP + (size_t)n * CBASE * PIX;

    float local = 0.0f;
    for (int pix = threadIdx.x; pix < PIX; pix += blockDim.x) {
        int oy = pix / RES, ox = pix - oy * RES;
        float gx = x1 + (ox + 0.5f) * bw;
        float gy = y1 + (oy + 0.5f) * bh;
        bool valid = (gy >= -1.0f) & (gy <= (float)HF) & (gx >= -1.0f) & (gx <= (float)WF);
        float yc = fminf(fmaxf(gy, 0.0f), (float)(HF - 1));
        float xc = fminf(fmaxf(gx, 0.0f), (float)(WF - 1));
        int y0 = (int)floorf(yc), x0 = (int)floorf(xc);
        int y1i = min(y0 + 1, HF - 1), x1i = min(x0 + 1, WF - 1);
        float ly = yc - (float)y0, lx = xc - (float)x0;
        float w00 = (1.0f - ly) * (1.0f - lx), w01 = (1.0f - ly) * lx;
        float w10 = ly * (1.0f - lx),          w11 = ly * lx;
        int i00 = y0 * WF + x0, i01 = y0 * WF + x1i;
        int i10 = y1i * WF + x0, i11 = y1i * WF + x1i;

        float pred = 0.0f;
#pragma unroll
        for (int c = 0; c < CBASE; ++c) {
            const float* f = salb + (size_t)c * HW;
            float rv = valid ? (f[i00]*w00 + f[i01]*w01 + f[i10]*w10 + f[i11]*w11) : 0.0f;
            float cu = cupn[(size_t)c * PIX + ox * RES + oy];
            pred += sigmoidf_(rv) * sigmoidf_(cu);
        }
        float gv = valid ? (mk[i00]*w00 + mk[i01]*w01 + mk[i10]*w10 + mk[i11]*w11) : 0.0f;
        float gt = (gv >= 0.5f) ? 1.0f : 0.0f;
        float p = fminf(fmaxf(pred, 1e-7f), 1.0f - 1e-7f);
        local += -(gt * logf(p) + (1.0f - gt) * log1pf(-p));
    }

    __shared__ float sdata[256];
    sdata[threadIdx.x] = local;
    __syncthreads();
    for (int s = 128; s > 0; s >>= 1) {
        if (threadIdx.x < s) sdata[threadIdx.x] += sdata[threadIdx.x + s];
        __syncthreads();
    }
    if (threadIdx.x == 0)
        atomicAdd(&acc[0], (sdata[0] * (1.0f / (float)PIX)) * ctval[n] * reg_mask[n]);
}

// ---------------- kernel 6: finalize scalar ----------------
__global__ void k_final(const float* acc, float* out) {
    out[0] = acc[0] / fmaxf(acc[1] * 0.25f, 1e-6f);
}

// ---------------- host launcher ----------------
extern "C" void kernel_launch(void* const* d_in, const int* in_sizes, int n_in,
                              void* d_out, int out_size, void* d_ws, size_t ws_size,
                              hipStream_t stream) {
    const float* saliency      = (const float*)d_in[0];   // (8,4,128,128)
    const float* shp           = (const float*)d_in[1];   // (8,784,128,128)
    const float* gtboxes       = (const float*)d_in[2];   // (8,64,4)
    const float* reg_mask      = (const float*)d_in[3];   // (8,64)
    const int*   ind           = (const int*)d_in[4];     // (8,64)
    const float* instance_mask = (const float*)d_in[5];   // (8,64,128,128)
    const float* center_target = (const float*)d_in[6];   // (8,80,128,128)
    const float* cat_mask      = (const float*)d_in[7];   // (8,64,80)
    float* out = (float*)d_out;

    char* ws = (char*)d_ws;
    float*    acc   = (float*)(ws + WS_ACC);
    float*    ctval = (float*)(ws + WS_CTVAL);
    _Float16* Bt    = (_Float16*)(ws + WS_BT);
    _Float16* Cg    = (_Float16*)(ws + WS_CG);
    _Float16* WT    = (_Float16*)(ws + WS_WT);
    float*    CUP   = (float*)(ws + WS_CUP);

    k_init<<<1, 256, 0, stream>>>(acc, Bt);
    k_ctval<<<2, 256, 0, stream>>>(center_target, cat_mask, ind, ctval, acc);
    k_gather<<<(M1ROWS + 255) / 256, 256, 0, stream>>>(shp, ind, Cg);
    k_stage1<<<M1TILES * NTILES / 8, 256, 0, stream>>>(Cg, Bt, WT);
    k_stage2<<<M2TILES * NTILES / 8, 256, 0, stream>>>(WT, Bt, CUP);
    k_loss<<<NB, 256, 0, stream>>>(saliency, gtboxes, reg_mask, instance_mask,
                                   ctval, CUP, acc);
    k_final<<<1, 1, 0, stream>>>(acc, out);
}
